// CrossDeformableAttention_31147102831015
// MI455X (gfx1250) — matrix operand order repaired and checked by reading, not compiled, not run on hardware
//
#include <hip/hip_runtime.h>
#include <hip/hip_bf16.h>

typedef __attribute__((ext_vector_type(16))) __bf16 v16bf;
typedef __attribute__((ext_vector_type(8)))  float  v8f;

#define HEADS   8
#define POINTS  4
#define BB_     4
#define CC_     256
#define WDIM    96
#define HDIM    96
#define NQ      (WDIM*HDIM)        /* 9216  */
#define M_TOT   (BB_*NQ)           /* 36864 */
#define N1      352                /* 256 value + 64 offsets + 32 attn logits */
#define N1PAD   384                /* padded to a multiple of 64 for 4-tile strips */

// ---------------------------------------------------------------------------
// WMMA fragment loaders (bf16, 16x16x32, wave32)
// ---------------------------------------------------------------------------
__device__ __forceinline__ v16bf load_a_frag(const __bf16* rowp, int k0, int half) {
    // A 16x32 bf16: half0 -> K=[k0,k0+8) U [k0+16,k0+24); half1 -> +8
    const __bf16* p = rowp + k0 + half * 8;
    v16bf f;
    ((uint4*)&f)[0] = *(const uint4*)(p);
    ((uint4*)&f)[1] = *(const uint4*)(p + 16);
    return f;
}
__device__ __forceinline__ v16bf load_b_frag(const __bf16* rowp, int k0, int half) {
    // B 32x16 bf16 (N-major storage): half0 -> K=[k0,k0+8) U [k0+16,k0+24); half1 -> +8
    const __bf16* p = rowp + k0 + half * 8;
    v16bf f;
    ((uint4*)&f)[0] = *(const uint4*)(p);
    ((uint4*)&f)[1] = *(const uint4*)(p + 16);
    return f;
}

// ---------------------------------------------------------------------------
// Cooperative GEMM block: 8 waves x (32x64) = 256x64 tile, K = 256.
// B panel (64 rows x 256 K, bf16, N-major) staged once in LDS (32 KB).
// Per k-step: 2 A frags (global) + 4 B frags (LDS) -> 8 WMMAs.
// acc[i][j] = 16x16 tile at rows m0+i*16, cols j*16.
// ---------------------------------------------------------------------------
__device__ __forceinline__ void stage_b_panel(__bf16* Bs, const __bf16* Bt_panel, int tid) {
#pragma unroll
    for (int it = 0; it < 8; ++it) {
        int idx = (it * 256 + tid) * 8;     // 8 bf16 = 16 B per chunk
        *(uint4*)(Bs + idx) = *(const uint4*)(Bt_panel + idx);
    }
    __syncthreads();
}

__device__ __forceinline__ void wmma_strip_2x4(const __bf16* A, const __bf16* Bs,
                                               int m0, int lane, v8f acc[2][4]) {
    const int half = lane >> 4;
    const int r    = lane & 15;
    const __bf16* arow0 = A + (size_t)(m0      + r) * 256;
    const __bf16* arow1 = A + (size_t)(m0 + 16 + r) * 256;
#pragma unroll
    for (int k0 = 0; k0 < 256; k0 += 32) {
        v16bf a0 = load_a_frag(arow0, k0, half);
        v16bf a1 = load_a_frag(arow1, k0, half);
#pragma unroll
        for (int j = 0; j < 4; ++j) {
            v16bf b = load_b_frag(Bs + (j * 16 + r) * 256, k0, half);
            acc[0][j] = __builtin_amdgcn_wmma_f32_16x16x32_bf16(
                            false, a0, false, b, (short)0, acc[0][j], false, false);
            acc[1][j] = __builtin_amdgcn_wmma_f32_16x16x32_bf16(
                            false, a1, false, b, (short)0, acc[1][j], false, false);
        }
    }
}

// ---------------------------------------------------------------------------
// Packing kernels
// ---------------------------------------------------------------------------
__global__ void pack_wcat(const float* __restrict__ Wv, const float* __restrict__ Woff,
                          const float* __restrict__ Wattn, __bf16* __restrict__ Wt1) {
    int i = blockIdx.x * blockDim.x + threadIdx.x;      // over 384*256
    if (i >= N1PAD * 256) return;
    int n = i / 256, k = i % 256;
    float v;
    if (n < 256)      v = Wv[(size_t)k * 256 + n];
    else if (n < 320) v = Woff[(size_t)k * 64 + (n - 256)];
    else if (n < N1)  v = Wattn[(size_t)k * 32 + (n - 320)];
    else              v = 0.f;                          // zero pad rows 352..383
    Wt1[(size_t)n * 256 + k] = (__bf16)v;               // N-major for B frags
}

__global__ void pack_wT(const float* __restrict__ W, __bf16* __restrict__ Wt) {
    int i = blockIdx.x * blockDim.x + threadIdx.x;      // over 256*256
    if (i >= 256 * 256) return;
    int n = i / 256, k = i % 256;
    Wt[(size_t)n * 256 + k] = (__bf16)W[(size_t)k * 256 + n];
}

// x[b][q][ch] = value[b][ch][wi][hi], q = wi*96 + hi  -> bf16 activation matrix
__global__ void pack_x(const float* __restrict__ value, __bf16* __restrict__ xb) {
    int i = blockIdx.x * blockDim.x + threadIdx.x;      // over M_TOT*256
    if (i >= M_TOT * 256) return;
    int m = i / 256, ch = i % 256;
    int b = m / NQ, q = m % NQ, wi = q / HDIM, hi = q % HDIM;
    xb[i] = (__bf16)value[(((size_t)b * CC_ + ch) * WDIM + wi) * HDIM + hi];
}

// ---------------------------------------------------------------------------
// GEMM 1: [M_TOT x 256] x [256 x 384(pad)]  ->  buf1 (fp32, v|off|attn concat)
// grid.x = (M_TOT/256) * (N1PAD/64)
// ---------------------------------------------------------------------------
__global__ void gemm_fused_voa(const __bf16* __restrict__ Xb, const __bf16* __restrict__ Wt1,
                               const float* __restrict__ bv, const float* __restrict__ boff,
                               const float* __restrict__ battn, float* __restrict__ buf1) {
    __shared__ __bf16 Bs[64 * 256];
    const int ngrp = N1PAD / 64;                        // 6
    int g    = blockIdx.x % ngrp;
    int mblk = blockIdx.x / ngrp;
    int tid  = threadIdx.x;
    int lane = tid & 31;
    int m0   = mblk * 256 + (tid >> 5) * 32;

    stage_b_panel(Bs, Wt1 + (size_t)(g * 64) * 256, tid);

    v8f acc[2][4] = {};
    wmma_strip_2x4(Xb, Bs, m0, lane, acc);

    int half = lane >> 4;
#pragma unroll
    for (int j = 0; j < 4; ++j) {
        int n = g * 64 + j * 16 + (lane & 15);
        if (n < N1) {                                   // skip zero-pad columns
            float bias = (n < 256) ? bv[n] : (n < 320 ? boff[n - 256] : battn[n - 320]);
#pragma unroll
            for (int i = 0; i < 2; ++i)
#pragma unroll
                for (int r = 0; r < 8; ++r) {
                    int m = m0 + i * 16 + r + half * 8;
                    buf1[(size_t)m * N1 + n] = acc[i][j][r] + bias;
                }
        }
    }
}

// ---------------------------------------------------------------------------
// Softmax + bilinear deformable sampling. One wave per (b,q,head); lane = channel.
// ---------------------------------------------------------------------------
__device__ __forceinline__ float fetchv(const float* vbase, int X, int Y) {
    if ((unsigned)X >= (unsigned)WDIM || (unsigned)Y >= (unsigned)HDIM) return 0.f;
    return vbase[(size_t)(Y * WDIM + X) * N1];
}

__global__ void deform_sample(const float* __restrict__ buf1, __bf16* __restrict__ outb) {
    int wid  = (int)((blockIdx.x * blockDim.x + threadIdx.x) >> 5);
    int lane = threadIdx.x & 31;
    if (wid >= BB_ * NQ * HEADS) return;
    int head = wid & (HEADS - 1);
    int q    = (wid >> 3) % NQ;
    int b    = wid / (NQ * HEADS);

    const float* row  = buf1 + (size_t)(b * NQ + q) * N1;
    const float* offp = row + 256 + head * (POINTS * 2);
    const float* atp  = row + 320 + head * POINTS;

    float l0 = atp[0], l1 = atp[1], l2 = atp[2], l3 = atp[3];
    float mx = fmaxf(fmaxf(l0, l1), fmaxf(l2, l3));
    float e0 = __expf(l0 - mx), e1 = __expf(l1 - mx);
    float e2 = __expf(l2 - mx), e3 = __expf(l3 - mx);
    float inv = 1.f / (e0 + e1 + e2 + e3);
    float aw[POINTS] = {e0 * inv, e1 * inv, e2 * inv, e3 * inv};

    int wi = q / HDIM, hi = q % HDIM;
    float refx = (float)wi * (1.0f / 95.0f);
    float refy = (float)hi * (1.0f / 95.0f);

    const float* vbase = buf1 + (size_t)b * NQ * N1 + head * 32 + lane;
    float acc = 0.f;
#pragma unroll
    for (int p = 0; p < POINTS; ++p) {
        float lx = refx + offp[2 * p]     * (1.0f / 96.0f);
        float ly = refy + offp[2 * p + 1] * (1.0f / 96.0f);
        float X = lx * 96.f - 0.5f;                 // (((2*loc-1)+1)*W - 1)/2
        float Y = ly * 96.f - 0.5f;
        float xf = floorf(X), yf = floorf(Y);
        float wx = X - xf, wy = Y - yf;
        int x0 = (int)xf, y0 = (int)yf;
        float v00 = fetchv(vbase, x0,     y0);
        float v10 = fetchv(vbase, x0 + 1, y0);
        float v01 = fetchv(vbase, x0,     y0 + 1);
        float v11 = fetchv(vbase, x0 + 1, y0 + 1);
        float s = v00 * (1.f - wx) * (1.f - wy) + v10 * wx * (1.f - wy)
                + v01 * (1.f - wx) * wy         + v11 * wx * wy;
        acc += aw[p] * s;
    }
    outb[(size_t)(b * NQ + q) * 256 + head * 32 + lane] = (__bf16)acc;
}

// ---------------------------------------------------------------------------
// GEMM 2: sampled @ W_proj1 + b_proj1 + residual(x from value)  -> tmp1 (bf16)
// grid.x = (M_TOT/256) * 4
// ---------------------------------------------------------------------------
__global__ void gemm_proj1(const __bf16* __restrict__ Ab, const __bf16* __restrict__ Wt,
                           const float* __restrict__ bias, const float* __restrict__ value,
                           __bf16* __restrict__ tmp1) {
    __shared__ __bf16 Bs[64 * 256];
    int g    = blockIdx.x & 3;
    int mblk = blockIdx.x >> 2;
    int tid  = threadIdx.x;
    int lane = tid & 31;
    int m0   = mblk * 256 + (tid >> 5) * 32;

    stage_b_panel(Bs, Wt + (size_t)(g * 64) * 256, tid);

    v8f acc[2][4] = {};
    wmma_strip_2x4(Ab, Bs, m0, lane, acc);

    int half = lane >> 4;
#pragma unroll
    for (int j = 0; j < 4; ++j) {
        int n = g * 64 + j * 16 + (lane & 15);
        float bn = bias[n];
#pragma unroll
        for (int i = 0; i < 2; ++i)
#pragma unroll
            for (int r = 0; r < 8; ++r) {
                int m = m0 + i * 16 + r + half * 8;
                int b = m / NQ, q = m % NQ, wi = q / HDIM, hi = q % HDIM;
                float res = value[(((size_t)b * CC_ + n) * WDIM + wi) * HDIM + hi];
                tmp1[(size_t)m * 256 + n] = (__bf16)(acc[i][j][r] + bn + res);
            }
    }
}

// ---------------------------------------------------------------------------
// GEMM 3: tmp1 @ W_proj2 + b_proj2, scattered to NCHW fp32 output.
// Per lane the 8 D-rows are 8 consecutive hi -> contiguous stores.
// ---------------------------------------------------------------------------
__global__ void gemm_proj2(const __bf16* __restrict__ Ab, const __bf16* __restrict__ Wt,
                           const float* __restrict__ bias, float* __restrict__ out) {
    __shared__ __bf16 Bs[64 * 256];
    int g    = blockIdx.x & 3;
    int mblk = blockIdx.x >> 2;
    int tid  = threadIdx.x;
    int lane = tid & 31;
    int m0   = mblk * 256 + (tid >> 5) * 32;

    stage_b_panel(Bs, Wt + (size_t)(g * 64) * 256, tid);

    v8f acc[2][4] = {};
    wmma_strip_2x4(Ab, Bs, m0, lane, acc);

    int half = lane >> 4;
#pragma unroll
    for (int j = 0; j < 4; ++j) {
        int n = g * 64 + j * 16 + (lane & 15);
        float bn = bias[n];
#pragma unroll
        for (int i = 0; i < 2; ++i)
#pragma unroll
            for (int r = 0; r < 8; ++r) {
                int m = m0 + i * 16 + r + half * 8;
                int b = m / NQ, q = m % NQ, wi = q / HDIM, hi = q % HDIM;
                out[(((size_t)b * CC_ + n) * WDIM + wi) * HDIM + hi] = acc[i][j][r] + bn;
            }
    }
}

// ---------------------------------------------------------------------------
// Host launcher
// ---------------------------------------------------------------------------
extern "C" void kernel_launch(void* const* d_in, const int* in_sizes, int n_in,
                              void* d_out, int out_size, void* d_ws, size_t ws_size,
                              hipStream_t stream) {
    (void)in_sizes; (void)n_in; (void)out_size; (void)ws_size;

    const float* value  = (const float*)d_in[0];
    const float* Wv     = (const float*)d_in[1];
    const float* bv     = (const float*)d_in[2];
    const float* Woff   = (const float*)d_in[3];
    const float* boff   = (const float*)d_in[4];
    const float* Wattn  = (const float*)d_in[5];
    const float* battn  = (const float*)d_in[6];
    const float* Wp1    = (const float*)d_in[7];
    const float* bp1    = (const float*)d_in[8];
    const float* Wp2    = (const float*)d_in[9];
    const float* bp2    = (const float*)d_in[10];
    float* out          = (float*)d_out;

    // Workspace carve-up (256B aligned)
    char*  ws  = (char*)d_ws;
    size_t off = 0;
    auto carve = [&](size_t bytes) -> char* {
        char* p = ws + off;
        off = (off + bytes + 255) & ~(size_t)255;
        return p;
    };
    __bf16* xb    = (__bf16*)carve((size_t)M_TOT * 256 * sizeof(__bf16));   // 18.9 MB
    float*  buf1  = (float* )carve((size_t)M_TOT * N1  * sizeof(float));    // 51.9 MB
    __bf16* sampb = (__bf16*)carve((size_t)M_TOT * 256 * sizeof(__bf16));   // 18.9 MB
    __bf16* tmp1  = (__bf16*)carve((size_t)M_TOT * 256 * sizeof(__bf16));   // 18.9 MB
    __bf16* wt1   = (__bf16*)carve((size_t)N1PAD * 256 * sizeof(__bf16));
    __bf16* wt2   = (__bf16*)carve((size_t)256 * 256 * sizeof(__bf16));
    __bf16* wt3   = (__bf16*)carve((size_t)256 * 256 * sizeof(__bf16));

    const int TB = 256;   // 8 waves per block

    // 1) pack weights (N-major bf16) + activations
    hipLaunchKernelGGL(pack_wcat, dim3((N1PAD * 256 + TB - 1) / TB), dim3(TB), 0, stream,
                       Wv, Woff, Wattn, wt1);
    hipLaunchKernelGGL(pack_wT, dim3((256 * 256 + TB - 1) / TB), dim3(TB), 0, stream, Wp1, wt2);
    hipLaunchKernelGGL(pack_wT, dim3((256 * 256 + TB - 1) / TB), dim3(TB), 0, stream, Wp2, wt3);
    hipLaunchKernelGGL(pack_x,  dim3((M_TOT * 256 + TB - 1) / TB), dim3(TB), 0, stream, value, xb);

    // 2) fused value/offset/attn GEMM : (M_TOT/256) * (384/64) = 144*6 blocks
    hipLaunchKernelGGL(gemm_fused_voa, dim3((M_TOT / 256) * (N1PAD / 64)), dim3(TB), 0, stream,
                       xb, wt1, bv, boff, battn, buf1);

    // 3) softmax + bilinear sampling : one wave per (b,q,head)
    hipLaunchKernelGGL(deform_sample, dim3((BB_ * NQ * HEADS) * 32 / TB), dim3(TB), 0, stream,
                       buf1, sampb);

    // 4) proj1 + residual : 144*4 blocks
    hipLaunchKernelGGL(gemm_proj1, dim3((M_TOT / 256) * 4), dim3(TB), 0, stream,
                       sampb, wt2, bp1, value, tmp1);

    // 5) proj2 + NCHW transpose store : 144*4 blocks
    hipLaunchKernelGGL(gemm_proj2, dim3((M_TOT / 256) * 4), dim3(TB), 0, stream,
                       tmp1, wt3, bp2, out);
}
